// rlf_87153476371160
// MI455X (gfx1250) — matrix operand order repaired and checked by reading, not compiled, not run on hardware
//
#include <hip/hip_runtime.h>

// ---------------------------------------------------------------------------
// MI455X (gfx1250) fused hypernetwork pipeline.
//
// Roofline: ~17 GMAC total GEMM work, ~35 MB HBM traffic (1.5us @ 23.3TB/s)
// -> matrix-pipe bound. All matmuls run on v_wmma_f32_16x16x32_f16 using
// 2-term f16 error-splitting (AhBh + AhBl + AlBh, f32 accumulate) for
// fp32-class accuracy at f16-WMMA rates. The per-sample hypernetwork weights
// (theta, 680MB if materialized) are algebraically eliminated: each hyper
// layer becomes a standard GEMM whose A rows are outer products zaug x aaug
// (formed on the fly) and whose B matrix is a shared gather of W_edec/W_fdec
// (pre-packed once per launch into WMMA wave32 lane layout).
//
// Each wave computes MW=2 batch strips x NTW n-tiles: every B tile loaded
// feeds 2x the WMMAs (6 per B pair), halving B-side L2 traffic and giving
// 4 independent accumulation chains per wave for latency hiding.
// ---------------------------------------------------------------------------

typedef _Float16 v16h __attribute__((ext_vector_type(16)));
typedef float    v8f  __attribute__((ext_vector_type(8)));

#define BATCH   2048
#define E_TOTAL 83008
#define F_TOTAL 82946

__device__ __forceinline__ v8f wmma_acc(v16h a, v16h b, v8f c) {
    // D = A*B + C, 16x16x32 f16 inputs, f32 accumulate
    return __builtin_amdgcn_wmma_f32_16x16x32_f16(false, a, false, b,
                                                  (short)0, c, false, false);
}

// ---------------------------------------------------------------------------
// GEMM: C(2048 x N) = A(2048 x Kpad) * Bpacked(Kpad x N)  [+bias] [+epilogue]
//   IA == 0 : A read directly from f32 row-major buffer (lda given)
//   IA  > 0 : A[b, k*IA+i] = zaug[b,k] * aaug[b,i]  (logical K = 33*IA)
//   EPI: 0 = store(+bias), 1 = ELU(+bias), 2 = cos/sin -> d_out (cols 0,1)
// One wave per block; wave computes (MW*16) rows x (NTW*16) cols, loops K.
// Bh/Bl are pre-split f16 hi/lo in WMMA lane layout:
//   Bh[((kt*NTt + nt)*32 + lane)*16 + e] = B[kt*32 + (e/8)*16 + (lane/16)*8 + e%8,
//                                            nt*16 + lane%16]
// A fragment layout (16-bit A 16x32): lane holds row M=lane%16, K chunks
//   (lane/16)*8 .. +7 and +16 .. +23.
// ---------------------------------------------------------------------------
template<int IA, int MW, int NTW, int EPI>
__global__ __launch_bounds__(32) void gemm_ws(
    const float* __restrict__ A, const float* __restrict__ zaug, int lda,
    const _Float16* __restrict__ Bh, const _Float16* __restrict__ Bl,
    int Kpad, int NTt, float* __restrict__ out, int ldo,
    const float* __restrict__ bias)
{
    const int lane   = threadIdx.x;            // 0..31
    const int strip0 = blockIdx.x * MW;        // first 16-row batch strip
    const int ntBase = blockIdx.y * NTW;       // first 16-col tile
    const int hgrp   = lane >> 4;              // 0 or 1
    const int mrow   = lane & 15;              // A-row within strip (A layout)
    const int h8     = hgrp * 8;

    v8f acc[MW][NTW];
#pragma unroll
    for (int s = 0; s < MW; ++s)
#pragma unroll
        for (int w = 0; w < NTW; ++w)
#pragma unroll
            for (int r = 0; r < 8; ++r) acc[s][w][r] = 0.0f;

    const float* arow[MW];
    const float* zrow[MW];
#pragma unroll
    for (int s = 0; s < MW; ++s) {
        int brow = (strip0 + s) * 16 + mrow;
        arow[s] = A + (size_t)brow * lda;
        zrow[s] = (IA > 0) ? (zaug + (size_t)brow * 33) : nullptr;
    }
    const int KT = Kpad >> 5;

    for (int kt = 0; kt < KT; ++kt) {
        // ---- build A fragments (hi/lo split f16), one per batch strip ---
        v16h ah[MW], al[MW];
        const int k1 = (kt << 5) + h8;         // K chunk1: k1..k1+7, chunk2: +16
#pragma unroll
        for (int s = 0; s < MW; ++s) {
            if constexpr (IA == 0) {
                const float4* p0 = (const float4*)(arow[s] + k1);
                const float4* p1 = (const float4*)(arow[s] + k1 + 16);
                float4 q0 = p0[0], q1 = p0[1], q2 = p1[0], q3 = p1[1];
#define SPL(j, v) { float _x = (v); _Float16 _h = (_Float16)_x; \
                    ah[s][j] = _h; al[s][j] = (_Float16)(_x - (float)_h); }
                SPL(0,q0.x)  SPL(1,q0.y)  SPL(2,q0.z)  SPL(3,q0.w)
                SPL(4,q1.x)  SPL(5,q1.y)  SPL(6,q1.z)  SPL(7,q1.w)
                SPL(8,q2.x)  SPL(9,q2.y)  SPL(10,q2.z) SPL(11,q2.w)
                SPL(12,q3.x) SPL(13,q3.y) SPL(14,q3.z) SPL(15,q3.w)
#undef SPL
            } else {
                constexpr int KLOG = 33 * IA;
#pragma unroll
                for (int j = 0; j < 16; ++j) {
                    int kk = k1 + (j < 8 ? j : 8 + j);   // j>=8 -> k1+16+(j-8)
                    int k  = kk / IA;                    // IA is constexpr
                    int i  = kk - k * IA;
                    bool ok = (kk < KLOG);
                    int kc = ok ? k : 0;                 // clamp: always in-bounds
                    int ic = ok ? i : 0;
                    float x = zrow[s][kc] * arow[s][ic];
                    x = ok ? x : 0.0f;
                    _Float16 h = (_Float16)x;
                    ah[s][j] = h;
                    al[s][j] = (_Float16)(x - (float)h);
                }
            }
        }
        // ---- B tiles: each pair feeds MW*3 WMMAs ------------------------
        if (kt + 1 < KT) {
            size_t pidx = ((size_t)((kt + 1) * NTt + ntBase) * 32 + lane) * 16;
            __builtin_prefetch(Bh + pidx, 0, 1);
            __builtin_prefetch(Bl + pidx, 0, 1);
        }
#pragma unroll
        for (int w = 0; w < NTW; ++w) {
            size_t idx = ((size_t)(kt * NTt + ntBase + w) * 32 + lane) * 16;
            v16h bh = *(const v16h*)(Bh + idx);
            v16h bl = *(const v16h*)(Bl + idx);
#pragma unroll
            for (int s = 0; s < MW; ++s) {
                acc[s][w] = wmma_acc(ah[s], bh, acc[s][w]);
                acc[s][w] = wmma_acc(ah[s], bl, acc[s][w]);
                acc[s][w] = wmma_acc(al[s], bh, acc[s][w]);
            }
        }
    }

    // ---- epilogue (C/D layout: col = lane%16, row = r + 8*(lane/16)) ----
#pragma unroll
    for (int s = 0; s < MW; ++s)
#pragma unroll
    for (int w = 0; w < NTW; ++w) {
        int coln = (ntBase + w) * 16 + mrow;
        float bv = 0.0f;
        if (EPI != 2 && bias) bv = bias[coln];
#pragma unroll
        for (int r = 0; r < 8; ++r) {
            int gr = (strip0 + s) * 16 + h8 + r;
            float v = acc[s][w][r];
            if constexpr (EPI == 2) {
                if (coln == 0) out[(size_t)gr * ldo + 0] = cosf(v);
                if (coln == 1) out[(size_t)gr * ldo + 1] = sinf(v);
            } else {
                v += bv;
                if constexpr (EPI == 1) v = (v > 0.0f) ? v : expm1f(v);
                out[(size_t)gr * ldo + coln] = v;
            }
        }
    }
}

// ---------------------------------------------------------------------------
// Pack a plain row-major (Klog x N) f32 matrix into hi/lo f16 WMMA layout.
// ---------------------------------------------------------------------------
__global__ void pack_plain(const float* __restrict__ W, _Float16* __restrict__ Bh,
                           _Float16* __restrict__ Bl, int Klog, int N, int KT)
{
    const int NTt = N >> 4;
    const long total = (long)KT * NTt * 512;
    long t = (long)blockIdx.x * blockDim.x + threadIdx.x;
    if (t >= total) return;
    int e    = (int)(t & 15);
    int L    = (int)((t >> 4) & 31);
    long rest = t >> 9;
    int nt   = (int)(rest % NTt);
    int kt   = (int)(rest / NTt);
    int kk   = kt * 32 + ((e >> 3) << 4) + ((L >> 4) << 3) + (e & 7);
    int n    = nt * 16 + (L & 15);
    float v  = (kk < Klog) ? W[(size_t)kk * N + n] : 0.0f;
    _Float16 h = (_Float16)v;
    Bh[t] = h;
    Bl[t] = (_Float16)(v - (float)h);
}

// ---------------------------------------------------------------------------
// Pack a hypernetwork B matrix: logical K index kk = k*IA + i (IA = IN+1).
//   i <  IN : row from W_dec weights  (theta offset Woff, layer out = Nreal)
//   i == IN : bias row                (theta offset Boff)
//   k <  32 : from W_dec[k, idx]; k == 32 : from b_dec[idx]
// ---------------------------------------------------------------------------
__global__ void pack_hyper(const float* __restrict__ Wd, const float* __restrict__ bd,
                           _Float16* __restrict__ Bh, _Float16* __restrict__ Bl,
                           int Woff, int Boff, int IN, int Nreal, int Npad,
                           int KT, int dec_total)
{
    const int NTt = Npad >> 4;
    const long total = (long)KT * NTt * 512;
    long t = (long)blockIdx.x * blockDim.x + threadIdx.x;
    if (t >= total) return;
    int e    = (int)(t & 15);
    int L    = (int)((t >> 4) & 31);
    long rest = t >> 9;
    int nt   = (int)(rest % NTt);
    int kt   = (int)(rest / NTt);
    int kk   = kt * 32 + ((e >> 3) << 4) + ((L >> 4) << 3) + (e & 7);
    int n    = nt * 16 + (L & 15);
    const int IA   = IN + 1;
    const int KLOG = 33 * IA;
    float v = 0.0f;
    if (kk < KLOG && n < Nreal) {
        int k = kk / IA;
        int i = kk - k * IA;
        int idx = (i < IN) ? (Woff + i * Nreal + n) : (Boff + n);
        v = (k < 32) ? Wd[(size_t)k * dec_total + idx] : bd[idx];
    }
    _Float16 h = (_Float16)v;
    Bh[t] = h;
    Bl[t] = (_Float16)(v - (float)h);
}

// ---------------------------------------------------------------------------
// Init: Saug/Gaug (states split + trailing 1) and all augmented-1 columns.
// ---------------------------------------------------------------------------
__global__ void init_aux(const float* __restrict__ states,
                         float* Saug, float* Gaug, float* zaug,
                         float* a1S, float* a1G, float* a2S, float* a2G,
                         float* f1, float* f2)
{
    int b = blockIdx.x * blockDim.x + threadIdx.x;
    if (b >= BATCH) return;
    Saug[b * 3 + 0] = states[b * 4 + 0];
    Saug[b * 3 + 1] = states[b * 4 + 1];
    Saug[b * 3 + 2] = 1.0f;
    Gaug[b * 3 + 0] = states[b * 4 + 2];
    Gaug[b * 3 + 1] = states[b * 4 + 3];
    Gaug[b * 3 + 2] = 1.0f;
    zaug[b * 33 + 32]  = 1.0f;
    a1S[b * 257 + 256] = 1.0f;
    a1G[b * 257 + 256] = 1.0f;
    a2S[b * 257 + 256] = 1.0f;
    a2G[b * 257 + 256] = 1.0f;
    f1 [b * 257 + 256] = 1.0f;
    f2 [b * 257 + 256] = 1.0f;
}

__global__ void make_dz(const float* __restrict__ eS, const float* __restrict__ eG,
                        float* __restrict__ dz)
{
    int t = blockIdx.x * blockDim.x + threadIdx.x;
    if (t >= BATCH * 65) return;
    int b = t / 65, i = t - b * 65;
    dz[t] = (i < 64) ? (eG[b * 64 + i] - eS[b * 64 + i]) : 1.0f;
}

// ---------------------------------------------------------------------------
extern "C" void kernel_launch(void* const* d_in, const int* in_sizes, int n_in,
                              void* d_out, int out_size, void* d_ws, size_t ws_size,
                              hipStream_t stream)
{
    (void)in_sizes; (void)n_in; (void)out_size; (void)ws_size;
    const float* maps   = (const float*)d_in[0];
    const float* states = (const float*)d_in[1];
    const float* W_enc1 = (const float*)d_in[2];
    const float* b_enc1 = (const float*)d_in[3];
    const float* W_enc2 = (const float*)d_in[4];
    const float* b_enc2 = (const float*)d_in[5];
    const float* W_enc3 = (const float*)d_in[6];
    const float* b_enc3 = (const float*)d_in[7];
    const float* W_bn   = (const float*)d_in[8];
    const float* b_bn   = (const float*)d_in[9];
    const float* W_edec = (const float*)d_in[10];
    const float* b_edec = (const float*)d_in[11];
    const float* W_fdec = (const float*)d_in[12];
    const float* b_fdec = (const float*)d_in[13];
    float* outp = (float*)d_out;

    char* base = (char*)d_ws;
    size_t off = 0;
    auto alloc  = [&](size_t nbytes) -> char* {
        size_t c = (off + 255) & ~(size_t)255; off = c + nbytes; return base + c;
    };
    auto allocH = [&](size_t halves) { return (_Float16*)alloc(halves * 2); };
    auto allocF = [&](size_t floats) { return (float*)alloc(floats * 4); };

    // packed B matrices (hi/lo pairs), sizes = Kpad * Npad halves
    _Float16 *pe1h = allocH(131072),  *pe1l = allocH(131072);   // enc1 1024x128
    _Float16 *pe2h = allocH(16384),   *pe2l = allocH(16384);    // enc2 128x128
    _Float16 *pe3h = allocH(16384),   *pe3l = allocH(16384);    // enc3 128x128
    _Float16 *pbnh = allocH(4096),    *pbnl = allocH(4096);     // bn   128x32
    _Float16 *pE1h = allocH(32768),   *pE1l = allocH(32768);    // E L1 128x256
    _Float16 *pE2h = allocH(2179072), *pE2l = allocH(2179072);  // E L2 8512x256
    _Float16 *pE3h = allocH(544768),  *pE3l = allocH(544768);   // E L3 8512x64
    _Float16 *pF1h = allocH(557056),  *pF1l = allocH(557056);   // F L1 2176x256
    _Float16 *pF2h = allocH(2179072), *pF2l = allocH(2179072);  // F L2 8512x256
    _Float16 *pF3h = allocH(136192),  *pF3l = allocH(136192);   // F L3 8512x16
    // f32 activations
    float *e1   = allocF(BATCH * 128), *e2 = allocF(BATCH * 128), *e3 = allocF(BATCH * 128);
    float *zaug = allocF(BATCH * 33);
    float *Saug = allocF(BATCH * 3),   *Gaug = allocF(BATCH * 3);
    float *a1S  = allocF(BATCH * 257), *a1G = allocF(BATCH * 257);
    float *a2S  = allocF(BATCH * 257), *a2G = allocF(BATCH * 257);
    float *eS   = allocF(BATCH * 64),  *eG  = allocF(BATCH * 64);
    float *dz   = allocF(BATCH * 65);
    float *f1   = allocF(BATCH * 257), *f2  = allocF(BATCH * 257);

    auto blocksFor = [](long total) { return (unsigned)((total + 255) / 256); };

    // ---- packing (independent) -----------------------------------------
    pack_plain<<<blocksFor((long)32 * 8 * 512), 256, 0, stream>>>(W_enc1, pe1h, pe1l, 1024, 128, 32);
    pack_plain<<<blocksFor((long)4 * 8 * 512), 256, 0, stream>>>(W_enc2, pe2h, pe2l, 128, 128, 4);
    pack_plain<<<blocksFor((long)4 * 8 * 512), 256, 0, stream>>>(W_enc3, pe3h, pe3l, 128, 128, 4);
    pack_plain<<<blocksFor((long)4 * 2 * 512), 256, 0, stream>>>(W_bn,   pbnh, pbnl, 128, 32, 4);
    // theta_e layout: W1[0,512) b1[512,768) W2[768,66304) b2[66304,66560)
    //                 W3[66560,82944) b3[82944,83008)
    pack_hyper<<<blocksFor((long)4 * 16 * 512),   256, 0, stream>>>(W_edec, b_edec, pE1h, pE1l, 0,     512,   2,   256, 256, 4,   E_TOTAL);
    pack_hyper<<<blocksFor((long)266 * 16 * 512), 256, 0, stream>>>(W_edec, b_edec, pE2h, pE2l, 768,   66304, 256, 256, 256, 266, E_TOTAL);
    pack_hyper<<<blocksFor((long)266 * 4 * 512),  256, 0, stream>>>(W_edec, b_edec, pE3h, pE3l, 66560, 82944, 256, 64,  64,  266, E_TOTAL);
    // theta_f layout: W1[0,16384) b1[16384,16640) W2[16640,82176) b2[82176,82432)
    //                 W3[82432,82944) b3[82944,82946)
    pack_hyper<<<blocksFor((long)68 * 16 * 512),  256, 0, stream>>>(W_fdec, b_fdec, pF1h, pF1l, 0,     16384, 64,  256, 256, 68,  F_TOTAL);
    pack_hyper<<<blocksFor((long)266 * 16 * 512), 256, 0, stream>>>(W_fdec, b_fdec, pF2h, pF2l, 16640, 82176, 256, 256, 256, 266, F_TOTAL);
    pack_hyper<<<blocksFor((long)266 * 1 * 512),  256, 0, stream>>>(W_fdec, b_fdec, pF3h, pF3l, 82432, 82944, 256, 2,   16,  266, F_TOTAL);
    init_aux<<<BATCH / 256, 256, 0, stream>>>(states, Saug, Gaug, zaug, a1S, a1G, a2S, a2G, f1, f2);

    const int SB = BATCH / 32;  // 64 blocks of MW=2 batch strips

    // ---- encoder -------------------------------------------------------
    gemm_ws<0, 2, 2, 1><<<dim3(SB, 4), 32, 0, stream>>>(maps, nullptr, 1024, pe1h, pe1l, 1024, 8, e1, 128, b_enc1);
    gemm_ws<0, 2, 2, 1><<<dim3(SB, 4), 32, 0, stream>>>(e1,   nullptr, 128,  pe2h, pe2l, 128,  8, e2, 128, b_enc2);
    gemm_ws<0, 2, 2, 1><<<dim3(SB, 4), 32, 0, stream>>>(e2,   nullptr, 128,  pe3h, pe3l, 128,  8, e3, 128, b_enc3);
    gemm_ws<0, 2, 2, 0><<<dim3(SB, 1), 32, 0, stream>>>(e3,   nullptr, 128,  pbnh, pbnl, 128,  2, zaug, 33, b_bn);

    // ---- hyper-E on S and G (A rows = zaug (x) aug-input) ---------------
    gemm_ws<3, 2, 2, 1>  <<<dim3(SB, 8), 32, 0, stream>>>(Saug, zaug, 3,   pE1h, pE1l, 128,  16, a1S, 257, nullptr);
    gemm_ws<3, 2, 2, 1>  <<<dim3(SB, 8), 32, 0, stream>>>(Gaug, zaug, 3,   pE1h, pE1l, 128,  16, a1G, 257, nullptr);
    gemm_ws<257, 2, 2, 1><<<dim3(SB, 8), 32, 0, stream>>>(a1S,  zaug, 257, pE2h, pE2l, 8512, 16, a2S, 257, nullptr);
    gemm_ws<257, 2, 2, 1><<<dim3(SB, 8), 32, 0, stream>>>(a1G,  zaug, 257, pE2h, pE2l, 8512, 16, a2G, 257, nullptr);
    gemm_ws<257, 2, 2, 0><<<dim3(SB, 2), 32, 0, stream>>>(a2S,  zaug, 257, pE3h, pE3l, 8512, 4,  eS,  64,  nullptr);
    gemm_ws<257, 2, 2, 0><<<dim3(SB, 2), 32, 0, stream>>>(a2G,  zaug, 257, pE3h, pE3l, 8512, 4,  eG,  64,  nullptr);

    make_dz<<<blocksFor((long)BATCH * 65), 256, 0, stream>>>(eS, eG, dz);

    // ---- hyper-F on dz, trig epilogue to d_out --------------------------
    gemm_ws<65, 2, 2, 1> <<<dim3(SB, 8), 32, 0, stream>>>(dz, zaug, 65,  pF1h, pF1l, 2176, 16, f1, 257, nullptr);
    gemm_ws<257, 2, 2, 1><<<dim3(SB, 8), 32, 0, stream>>>(f1, zaug, 257, pF2h, pF2l, 8512, 16, f2, 257, nullptr);
    gemm_ws<257, 2, 1, 2><<<dim3(SB, 1), 32, 0, stream>>>(f2, zaug, 257, pF3h, pF3l, 8512, 1,  outp, 2, nullptr);
}
